// LSTM_6786048328528
// MI455X (gfx1250) — matrix operand order repaired and checked by reading, not compiled, not run on hardware
//
#include <hip/hip_runtime.h>
#include <hip/hip_bf16.h>

// Problem constants (from the reference)
#define HH       64     // hidden size
#define TT       64     // timesteps
#define FEATN    8      // input features (layer 0)
#define TOUTN    12     // output features
#define NL       5      // LSTM layers
#define GATES    256    // 4*H
#define SEQB     16     // sequences per block (one WMMA M-tile)
#define NSEQ     (64 * 128)

typedef __attribute__((ext_vector_type(16))) __bf16 v16bf;
typedef __attribute__((ext_vector_type(8)))  float  v8f;

__device__ __forceinline__ unsigned short f32_to_bf16_bits(float f) {
  union { float f; unsigned u; } a; a.f = f;
  unsigned r = a.u + 0x7FFFu + ((a.u >> 16) & 1u);   // round-to-nearest-even
  return (unsigned short)(r >> 16);
}
__device__ __forceinline__ float bf16_bits_to_f32(unsigned short s) {
  union { unsigned u; float f; } a; a.u = ((unsigned)s) << 16;
  return a.f;
}
__device__ __forceinline__ __bf16 bits_to_bf(unsigned short s) {
  union { unsigned short s; __bf16 b; } a; a.s = s; return a.b;
}
// one transcendental each
__device__ __forceinline__ float fast_sigmoid(float x) {
  return 1.0f / (1.0f + __expf(-x));
}
__device__ __forceinline__ float fast_tanh(float x) {
  float e = __expf(2.0f * x);
  return (e - 1.0f) / (e + 1.0f);
}

// B fragment (32x16 bf16) from LDS bf16-bits row-major [GATES][HH].
// Lane n = output column; lanes 0-15: K=k0..k0+15, lanes 16-31: K=k0+16..k0+31
// (16 contiguous u16 per lane -> 2x ds_load_b128).
__device__ __forceinline__ v16bf load_b_frag_lds(const unsigned short* __restrict__ Wb,
                                                 int row_base, int k0, int lane) {
  int n  = lane & 15;
  int ks = k0 + ((lane & 16) ? 16 : 0);
  const unsigned short* rp = Wb + (row_base + n) * HH + ks;
  v16bf v;
#pragma unroll
  for (int e = 0; e < 16; ++e) v[e] = bits_to_bf(rp[e]);
  return v;
}

// A fragment (16x32 bf16) from LDS row-major [16][64] bf16-bits.
// Lane m<16: K = k0..k0+15; lane m+16: K = k0+16..k0+31.
__device__ __forceinline__ v16bf load_a_frag_lds(const unsigned short* __restrict__ h,
                                                 int k0, int lane) {
  int m  = lane & 15;
  int kb = k0 + ((lane & 16) ? 16 : 0);
  const unsigned short* rp = h + m * HH;
  v16bf v;
#pragma unroll
  for (int e = 0; e < 16; ++e) {
    int k = kb + e;
    v[e] = bits_to_bf(rp[k]);
  }
  return v;
}

#define WMMA_BF16(A, B, C) \
  __builtin_amdgcn_wmma_f32_16x16x32_bf16(false, (A), false, (B), (short)0, (C), false, false)

__global__ __launch_bounds__(128, 1)
void lstm5_fused_wmma(const float* __restrict__ X,
                      const float* __restrict__ W_ih0, const float* __restrict__ W_hh0,
                      const float* __restrict__ b_ih0, const float* __restrict__ b_hh0,
                      const float* __restrict__ W_ih_r, const float* __restrict__ W_hh_r,
                      const float* __restrict__ b_ih_r, const float* __restrict__ b_hh_r,
                      const float* __restrict__ W_out, const float* __restrict__ b_out,
                      float* __restrict__ out) {
  // LDS: h-sequence ring (in-place across layers) + one reusable weight stage.
  __shared__ unsigned short hbuf[TT * SEQB * HH];   // 128 KB (bf16 bits)
  __shared__ unsigned short wbuf[GATES * HH];       //  32 KB (bf16 bits)

  const int tid   = threadIdx.x;
  const int lane  = tid & 31;
  const int wv    = tid >> 5;                // wave = hidden-unit slice [16w,16w+16)
  const int bn0   = blockIdx.x * SEQB;
  const int ucol  = wv * 16 + (lane & 15);   // hidden unit owned by this lane
  const int mhalf = (lane & 16) ? 8 : 0;     // sequence base for D rows

  // Prefill hbuf with the zero-padded layer-0 input: hbuf[t][seq][k<8] = X.
  for (int idx = tid; idx < TT * SEQB * HH; idx += blockDim.x) {
    int t   = idx / (SEQB * HH);
    int r   = idx - t * (SEQB * HH);
    int seq = r >> 6;
    int k   = r & 63;
    float v = 0.0f;
    if (k < FEATN) v = X[((size_t)(bn0 + seq) * TT + t) * FEATN + k];
    hbuf[idx] = f32_to_bf16_bits(v);
  }

  for (int layer = 0; layer < NL; ++layer) {
    const float* Wih; int ldw, kdim;
    const float* Whh; const float* bih; const float* bhh;
    if (layer == 0) {
      Wih = W_ih0; ldw = FEATN; kdim = FEATN;
      Whh = W_hh0; bih = b_ih0; bhh = b_hh0;
    } else {
      Wih = W_ih_r + (size_t)(layer - 1) * GATES * HH; ldw = HH; kdim = HH;
      Whh = W_hh_r + (size_t)(layer - 1) * GATES * HH;
      bih = b_ih_r + (size_t)(layer - 1) * GATES;
      bhh = b_hh_r + (size_t)(layer - 1) * GATES;
    }

    // Per-lane combined bias for the 4 gates of this lane's unit column.
    float bias[4];
#pragma unroll
    for (int g = 0; g < 4; ++g)
      bias[g] = bih[g * HH + ucol] + bhh[g * HH + ucol];

    // --- Stage W_hh into wbuf (rolled, coalesced), pull fragments to VGPRs ---
    for (int idx = tid; idx < GATES * HH; idx += blockDim.x)
      wbuf[idx] = f32_to_bf16_bits(Whh[idx]);          // W_hh is [256][64], ldw==HH
    __syncthreads();
    v16bf Bhh[4][2];
#pragma unroll
    for (int g = 0; g < 4; ++g)
#pragma unroll
      for (int kh = 0; kh < 2; ++kh)
        Bhh[g][kh] = load_b_frag_lds(wbuf, g * HH + wv * 16, kh * 32, lane);
    __syncthreads();   // everyone done reading wbuf before restage

    // --- Restage wbuf with W_ih (zero-padded K for layer 0) ---
    for (int idx = tid; idx < GATES * HH; idx += blockDim.x) {
      int row = idx >> 6, k = idx & 63;
      float v = (k < kdim) ? Wih[(size_t)row * ldw + k] : 0.0f;
      wbuf[idx] = f32_to_bf16_bits(v);
    }

    // Cell state in registers: creg[r] for (seq = mhalf+r, unit = ucol).
    float creg[8];
#pragma unroll
    for (int r = 0; r < 8; ++r) creg[r] = 0.0f;

    __syncthreads();   // wbuf(W_ih) staged; also covers hbuf prefill on layer 0

    // ---------------- t = 0 (peeled: no recurrent term) ----------------
    {
      v16bf ax0 = load_a_frag_lds(&hbuf[0], 0, lane);
      v16bf ax1 = load_a_frag_lds(&hbuf[0], 32, lane);
      __syncthreads();  // all reads of slot 0 done before it is overwritten

      v8f a0, a1, a2, a3;
#pragma unroll
      for (int r = 0; r < 8; ++r) { a0[r] = bias[0]; a1[r] = bias[1]; a2[r] = bias[2]; a3[r] = bias[3]; }
      {
        v16bf bi0 = load_b_frag_lds(wbuf, 0 * HH + wv * 16, 0, lane);
        v16bf bi1 = load_b_frag_lds(wbuf, 1 * HH + wv * 16, 0, lane);
        v16bf bi2 = load_b_frag_lds(wbuf, 2 * HH + wv * 16, 0, lane);
        v16bf bi3 = load_b_frag_lds(wbuf, 3 * HH + wv * 16, 0, lane);
        a0 = WMMA_BF16(ax0, bi0, a0);
        a1 = WMMA_BF16(ax0, bi1, a1);
        a2 = WMMA_BF16(ax0, bi2, a2);
        a3 = WMMA_BF16(ax0, bi3, a3);
      }
      {
        v16bf bi0 = load_b_frag_lds(wbuf, 0 * HH + wv * 16, 32, lane);
        v16bf bi1 = load_b_frag_lds(wbuf, 1 * HH + wv * 16, 32, lane);
        v16bf bi2 = load_b_frag_lds(wbuf, 2 * HH + wv * 16, 32, lane);
        v16bf bi3 = load_b_frag_lds(wbuf, 3 * HH + wv * 16, 32, lane);
        a0 = WMMA_BF16(ax1, bi0, a0);
        a1 = WMMA_BF16(ax1, bi1, a1);
        a2 = WMMA_BF16(ax1, bi2, a2);
        a3 = WMMA_BF16(ax1, bi3, a3);
      }
      unsigned short* hw = &hbuf[0];
#pragma unroll
      for (int r = 0; r < 8; ++r) {
        float ig = fast_sigmoid(a0[r]);
        float fg = fast_sigmoid(a1[r]);
        float gg = fast_tanh   (a2[r]);
        float og = fast_sigmoid(a3[r]);
        creg[r] = fg * creg[r] + ig * gg;
        float h = og * fast_tanh(creg[r]);
        hw[(mhalf + r) * HH + ucol] = f32_to_bf16_bits(h);
      }
      __syncthreads();  // slot 0 complete
    }

    // ---------------- t = 1 .. T-1 (branch-free steady state) ----------------
#pragma unroll 1
    for (int t = 1; t < TT; ++t) {
      const unsigned short* hx = &hbuf[t * SEQB * HH];
      const unsigned short* hp = &hbuf[(t - 1) * SEQB * HH];
      v16bf ax0 = load_a_frag_lds(hx, 0, lane);
      v16bf ax1 = load_a_frag_lds(hx, 32, lane);
      v16bf ah0 = load_a_frag_lds(hp, 0, lane);
      v16bf ah1 = load_a_frag_lds(hp, 32, lane);
      __syncthreads();  // everyone finished reading slot t before it is overwritten

      v8f a0, a1, a2, a3;
#pragma unroll
      for (int r = 0; r < 8; ++r) { a0[r] = bias[0]; a1[r] = bias[1]; a2[r] = bias[2]; a3[r] = bias[3]; }

      // K-half 0: issue Bih loads, cover their latency with register-operand
      // Bhh WMMAs, then consume the loads.
      {
        v16bf bi0 = load_b_frag_lds(wbuf, 0 * HH + wv * 16, 0, lane);
        v16bf bi1 = load_b_frag_lds(wbuf, 1 * HH + wv * 16, 0, lane);
        v16bf bi2 = load_b_frag_lds(wbuf, 2 * HH + wv * 16, 0, lane);
        v16bf bi3 = load_b_frag_lds(wbuf, 3 * HH + wv * 16, 0, lane);
        a0 = WMMA_BF16(ah0, Bhh[0][0], a0);
        a1 = WMMA_BF16(ah0, Bhh[1][0], a1);
        a2 = WMMA_BF16(ah0, Bhh[2][0], a2);
        a3 = WMMA_BF16(ah0, Bhh[3][0], a3);
        a0 = WMMA_BF16(ax0, bi0, a0);
        a1 = WMMA_BF16(ax0, bi1, a1);
        a2 = WMMA_BF16(ax0, bi2, a2);
        a3 = WMMA_BF16(ax0, bi3, a3);
      }
      // K-half 1: same pattern.
      {
        v16bf bi0 = load_b_frag_lds(wbuf, 0 * HH + wv * 16, 32, lane);
        v16bf bi1 = load_b_frag_lds(wbuf, 1 * HH + wv * 16, 32, lane);
        v16bf bi2 = load_b_frag_lds(wbuf, 2 * HH + wv * 16, 32, lane);
        v16bf bi3 = load_b_frag_lds(wbuf, 3 * HH + wv * 16, 32, lane);
        a0 = WMMA_BF16(ah1, Bhh[0][1], a0);
        a1 = WMMA_BF16(ah1, Bhh[1][1], a1);
        a2 = WMMA_BF16(ah1, Bhh[2][1], a2);
        a3 = WMMA_BF16(ah1, Bhh[3][1], a3);
        a0 = WMMA_BF16(ax1, bi0, a0);
        a1 = WMMA_BF16(ax1, bi1, a1);
        a2 = WMMA_BF16(ax1, bi2, a2);
        a3 = WMMA_BF16(ax1, bi3, a3);
      }

      // ---- LSTM cell update fully in registers; write h (bf16) to slot t ----
      unsigned short* hw = &hbuf[t * SEQB * HH];
#pragma unroll
      for (int r = 0; r < 8; ++r) {
        float ig = fast_sigmoid(a0[r]);
        float fg = fast_sigmoid(a1[r]);
        float gg = fast_tanh   (a2[r]);
        float og = fast_sigmoid(a3[r]);
        creg[r] = fg * creg[r] + ig * gg;
        float h = og * fast_tanh(creg[r]);
        hw[(mhalf + r) * HH + ucol] = f32_to_bf16_bits(h);
      }
      __syncthreads();  // slot t complete for step t+1 readers
    }
  }

  // Output head: out[seq] = h_{T-1} @ W_out^T + b_out  (tiny: plain VALU)
  for (int o = tid; o < SEQB * TOUTN; o += blockDim.x) {
    int seq = o / TOUTN, u = o - seq * TOUTN;
    const unsigned short* hr = &hbuf[((TT - 1) * SEQB + seq) * HH];
    const float* wr = W_out + u * HH;
    float s = b_out[u];
#pragma unroll
    for (int k = 0; k < HH; ++k) s += bf16_bits_to_f32(hr[k]) * wr[k];
    out[(size_t)(bn0 + seq) * TOUTN + u] = s;
  }
}

extern "C" void kernel_launch(void* const* d_in, const int* in_sizes, int n_in,
                              void* d_out, int out_size, void* d_ws, size_t ws_size,
                              hipStream_t stream) {
  // setup_inputs order:
  // 0 A_hat (unused), 1 X_joint, 2 X_motion (unused), 3 W_ih0, 4 W_hh0,
  // 5 b_ih0, 6 b_hh0, 7 W_ih_r, 8 W_hh_r, 9 b_ih_r, 10 b_hh_r, 11 W_out, 12 b_out
  const float* X      = (const float*)d_in[1];
  const float* W_ih0  = (const float*)d_in[3];
  const float* W_hh0  = (const float*)d_in[4];
  const float* b_ih0  = (const float*)d_in[5];
  const float* b_hh0  = (const float*)d_in[6];
  const float* W_ih_r = (const float*)d_in[7];
  const float* W_hh_r = (const float*)d_in[8];
  const float* b_ih_r = (const float*)d_in[9];
  const float* b_hh_r = (const float*)d_in[10];
  const float* W_out  = (const float*)d_in[11];
  const float* b_out  = (const float*)d_in[12];
  float* out = (float*)d_out;

  dim3 grid(NSEQ / SEQB);   // 512 blocks of 16 sequences
  dim3 block(128);          // 4 wave32 per block (one per 16-unit slice)
  lstm5_fused_wmma<<<grid, block, 0, stream>>>(X, W_ih0, W_hh0, b_ih0, b_hh0,
                                               W_ih_r, W_hh_r, b_ih_r, b_hh_r,
                                               W_out, b_out, out);
}
